// pointcloudNN_69887707841100
// MI455X (gfx1250) — matrix operand order repaired and checked by reading, not compiled, not run on hardware
//
#include <hip/hip_runtime.h>
#include <hip/hip_bf16.h>
#include <math.h>

#define NPTS   32768
#define KNBR   32
#define EEDGES (NPTS * KNBR)
#define PTS    1024

typedef __attribute__((ext_vector_type(16))) _Float16 v16h;
typedef __attribute__((ext_vector_type(8)))  _Float16 v8h;
typedef __attribute__((ext_vector_type(8)))  float    v8f;
typedef __attribute__((ext_vector_type(4)))  _Float16 v4h;

// ---------------------------------------------------------------------------
// Branch-free tanh: (e^{2x}-1)/(e^{2x}+1) with hw exp/rcp; clamp avoids inf.
// Accuracy ~1e-6 absolute — far below the f16 quantization we store into.
// ---------------------------------------------------------------------------
__device__ __forceinline__ float fast_tanh(float x) {
  float xc = fminf(fmaxf(x, -9.0f), 9.0f);
  float e  = __expf(2.0f * xc);                       // v_exp_f32 path
  return (e - 1.0f) * __builtin_amdgcn_rcpf(e + 1.0f); // v_rcp_f32
}

// ---------------------------------------------------------------------------
// WMMA helpers (CDNA5 16x16x32 f16 -> f32)
// ---------------------------------------------------------------------------
__device__ __forceinline__ v8f wmma_f16(v16h a, v16h b, v8f c) {
  return __builtin_amdgcn_wmma_f32_16x16x32_f16(false, a, false, b, (short)0, c,
                                                false, false);
}

// A fragment: 16x32 f16 tile from row-major X (lda elems), K offset k0.
__device__ __forceinline__ v16h load_a(const _Float16* X, int lda, int k0) {
  const int lane = threadIdx.x & 31;
  const int m    = lane & 15;
  const int base = (lane >> 4) * 8;
  const _Float16* row = X + m * lda + k0 + base;
  v16h a;
#pragma unroll
  for (int h = 0; h < 8; ++h) {
    a[h]     = row[h];        // K = base + h
    a[h + 8] = row[16 + h];   // K = base + 16 + h
  }
  return a;
}

// B fragment: 32x16 slice of W^T; W row-major [out][ld] f16.
__device__ __forceinline__ v16h load_b(const _Float16* W, int ld, int n0, int k0) {
  const int lane = threadIdx.x & 31;
  const int n    = lane & 15;
  const int kb   = k0 + (lane >> 4) * 8;
  const _Float16* col = W + (size_t)(n0 + n) * ld + kb;
  v16h b;
#pragma unroll
  for (int h = 0; h < 8; ++h) {
    b[h]     = col[h];
    b[h + 8] = col[16 + h];
  }
  return b;
}

// Store C/D tile (v8f) into row-major f16 out (ldo elems) at col n0.
__device__ __forceinline__ void store_tanh_lds(_Float16* out, int ldo, int n0, v8f d) {
  const int lane = threadIdx.x & 31;
  const int n  = lane & 15;
  const int mb = (lane >> 4) * 8;
#pragma unroll
  for (int r = 0; r < 8; ++r)
    out[(r + mb) * ldo + n0 + n] = (_Float16)fast_tanh(d[r]);
}

__device__ __forceinline__ void store_raw_lds(_Float16* out, int ldo, int n0, v8f d) {
  const int lane = threadIdx.x & 31;
  const int n  = lane & 15;
  const int mb = (lane >> 4) * 8;
#pragma unroll
  for (int r = 0; r < 8; ++r)
    out[(r + mb) * ldo + n0 + n] = (_Float16)d[r];
}

__device__ __forceinline__ v8f bias_init(const float* b, int n0) {
  const float bv = b[n0 + (threadIdx.x & 15)];
  v8f c;
#pragma unroll
  for (int r = 0; r < 8; ++r) c[r] = bv;
  return c;
}

// ---------------------------------------------------------------------------
// Kernel 0: weight f16 conversion (+ K-padding to x32) and pooled zeroing
// ---------------------------------------------------------------------------
__global__ void prep_kernel(const float* e1w, const float* e2w, const float* e3w,
                            const float* locw, const float* globw,
                            _Float16* wE1, _Float16* wE2, _Float16* wE3,
                            _Float16* wLoc, _Float16* wGlob, float* pooled) {
  const int tid = blockIdx.x * 256 + threadIdx.x;
  switch (blockIdx.y) {
    case 0:  // enc1: [64][3] -> [64][32] zero-padded
      if (tid < 64 * 32) {
        int r = tid >> 5, c = tid & 31;
        wE1[tid] = (_Float16)(c < 3 ? e1w[r * 3 + c] : 0.0f);
      }
      break;
    case 1:
      if (tid < 64 * 64) wE2[tid] = (_Float16)e2w[tid];
      break;
    case 2:
      if (tid < 128 * 64) wE3[tid] = (_Float16)e3w[tid];
      break;
    case 3:  // loc: [256][131] -> [256][160] zero-padded
      if (tid < 256 * 160) {
        int r = tid / 160, c = tid % 160;
        wLoc[tid] = (_Float16)(c < 131 ? locw[r * 131 + c] : 0.0f);
      }
      break;
    case 4:
      if (tid < 384 * 256) wGlob[tid] = (_Float16)globw[tid];
      break;
    case 5:
      if (tid < 32 * 384) pooled[tid] = 0.0f;  // re-zeroed every launch
      break;
  }
}

// ---------------------------------------------------------------------------
// Kernel 1: encoder MLP  pos[N][3] -> xEnc[N][128] f16   (WMMA, tanh)
// 128 threads = 4 waves, 16 points per wave, 64 points per workgroup.
// ---------------------------------------------------------------------------
__global__ void encoder_kernel(const float* pos,
                               const float* b1, const float* b2, const float* b3,
                               const _Float16* wE1, const _Float16* wE2,
                               const _Float16* wE3, _Float16* xEnc) {
  __shared__ _Float16 lds[4][16 * 32 + 16 * 64 + 16 * 64];
  __shared__ _Float16 outB[4][16 * 128];
  const int wave = threadIdx.x >> 5;
  const int lane = threadIdx.x & 31;
  _Float16* inP = &lds[wave][0];
  _Float16* hA  = inP + 16 * 32;
  _Float16* hB  = hA + 16 * 64;
  _Float16* oB  = &outB[wave][0];
  const int pt0 = (blockIdx.x * 4 + wave) * 16;

  // Stage pos tile [16][32] (K padded with zeros)
  for (int i = lane; i < 16 * 32; i += 32) {
    int m = i >> 5, c = i & 31;
    inP[i] = (_Float16)(c < 3 ? pos[(size_t)(pt0 + m) * 3 + c] : 0.0f);
  }
  __syncthreads();

  // Layer 1: [16x32] x [32x64]
#pragma unroll
  for (int nt = 0; nt < 4; ++nt) {
    const int n0 = nt * 16;
    v8f c = bias_init(b1, n0);
    c = wmma_f16(load_a(inP, 32, 0), load_b(wE1, 32, n0, 0), c);
    store_tanh_lds(hA, 64, n0, c);
  }
  __syncthreads();

  // Layer 2: [16x64] x [64x64]
#pragma unroll
  for (int nt = 0; nt < 4; ++nt) {
    const int n0 = nt * 16;
    v8f c = bias_init(b2, n0);
#pragma unroll
    for (int k0 = 0; k0 < 64; k0 += 32)
      c = wmma_f16(load_a(hA, 64, k0), load_b(wE2, 64, n0, k0), c);
    store_tanh_lds(hB, 64, n0, c);
  }
  __syncthreads();

  // Layer 3: [16x64] x [64x128] -> LDS tile, then coalesced b128 stores
#pragma unroll
  for (int nt = 0; nt < 8; ++nt) {
    const int n0 = nt * 16;
    v8f c = bias_init(b3, n0);
#pragma unroll
    for (int k0 = 0; k0 < 64; k0 += 32)
      c = wmma_f16(load_a(hB, 64, k0), load_b(wE3, 64, n0, k0), c);
    store_tanh_lds(oB, 128, n0, c);
  }
  __syncthreads();
  // 2048 halfs per wave -> 8 x (ds_load_b128 + global_store_b128) per lane
  _Float16* dst = xEnc + (size_t)pt0 * 128;
#pragma unroll
  for (int i = 0; i < 8; ++i) {
    const int off = (i * 32 + lane) * 8;
    *(v8h*)(dst + off) = *(const v8h*)(oB + off);
  }
}

// ---------------------------------------------------------------------------
// Kernel 2: init S with the self-loop contribution: S[i] = [x_i, 0,0,0, 1]
// S layout: [N][132]  (128 x-sum | 3 rel-sum | 1 count). 4-wide vectorized.
// ---------------------------------------------------------------------------
__global__ void initS_kernel(const _Float16* xEnc, float* S) {
  const int tid = blockIdx.x * 256 + threadIdx.x;  // N*33 float4 units
  if (tid >= NPTS * 33) return;
  const int i = tid / 33, c4 = tid % 33;
  float4 v;
  if (c4 < 32) {
    v4h x = *(const v4h*)(xEnc + (size_t)i * 128 + c4 * 4);
    v = make_float4((float)x[0], (float)x[1], (float)x[2], (float)x[3]);
  } else {
    v = make_float4(0.0f, 0.0f, 0.0f, 1.0f);  // rel-sum zeros + self-loop count
  }
  *(float4*)(S + (size_t)i * 132 + c4 * 4) = v;
}

// ---------------------------------------------------------------------------
// Kernel 3: edge scatter-add (linearity trick: sum features, not messages)
// One wave per edge: lane gathers 4 f16 channels of x[src], atomic-adds f32.
// ---------------------------------------------------------------------------
__global__ void scatter_kernel(const long long* ei, const float* pos,
                               const _Float16* xEnc, float* S) {
  const int e    = blockIdx.x * 8 + (threadIdx.x >> 5);
  const int lane = threadIdx.x & 31;
  const long long src = ei[e];
  const long long dst = ei[EEDGES + e];
  const v4h xv = *(const v4h*)(xEnc + (size_t)src * 128 + lane * 4);
  float* Srow = S + (size_t)dst * 132;
#pragma unroll
  for (int i = 0; i < 4; ++i)
    atomicAdd(&Srow[lane * 4 + i], (float)xv[i]);
  if (lane < 3)
    atomicAdd(&Srow[128 + lane],
              pos[(size_t)src * 3 + lane] - pos[(size_t)dst * 3 + lane]);
  if (lane == 3) atomicAdd(&Srow[131], 1.0f);
}

// ---------------------------------------------------------------------------
// Kernel 4: scatter-mean normalize -> Sh f16 [N][160] (zero-padded K)
// 4-wide vectorized: b128 f32 loads, b64 f16 stores.
// ---------------------------------------------------------------------------
__global__ void norm_kernel(const float* S, _Float16* Sh) {
  const int tid = blockIdx.x * 256 + threadIdx.x;  // N*40 v4h units
  if (tid >= NPTS * 40) return;
  const int i = tid / 40, c4 = tid % 40;
  const float* Srow = S + (size_t)i * 132;
  v4h o = {(_Float16)0.f, (_Float16)0.f, (_Float16)0.f, (_Float16)0.f};
  if (c4 < 33) {
    const float rinv = __builtin_amdgcn_rcpf(Srow[131]);  // cnt >= 1 always
    float4 s = *(const float4*)(Srow + c4 * 4);
    o[0] = (_Float16)(s.x * rinv);
    o[1] = (_Float16)(s.y * rinv);
    o[2] = (_Float16)(s.z * rinv);
    o[3] = (c4 < 32) ? (_Float16)(s.w * rinv) : (_Float16)0.f;  // col 131 pad
  }
  *(v4h*)(Sh + (size_t)i * 160 + c4 * 4) = o;
}

// ---------------------------------------------------------------------------
// Kernel 5: agg = Sh @ locW^T + locB ; x2 = agg @ globW^T + globB ;
//           per-cloud sum of x2 rows -> pooled[32][384] (x2 never hits HBM)
// ---------------------------------------------------------------------------
__global__ void aggpool_kernel(const _Float16* Sh, const _Float16* wLoc,
                               const float* locB, const _Float16* wGlob,
                               const float* globB, float* pooled) {
  __shared__ _Float16 aggL[64 * 256];  // 32 KB
  const int wave = threadIdx.x >> 5;
  const int lane = threadIdx.x & 31;
  const int pt0  = blockIdx.x * 64 + wave * 16;
  const _Float16* A = Sh + (size_t)pt0 * 160;
  _Float16* myAgg = aggL + wave * 16 * 256;

  // Phase 1: agg tile [16x256], K = 160
  for (int nt = 0; nt < 16; ++nt) {
    const int n0 = nt * 16;
    v8f c = bias_init(locB, n0);
#pragma unroll
    for (int k0 = 0; k0 < 160; k0 += 32)
      c = wmma_f16(load_a(A, 160, k0), load_b(wLoc, 160, n0, k0), c);
    store_raw_lds(myAgg, 256, n0, c);
  }
  __syncthreads();

  // Phase 2: x2 tile [16x384], K = 256, fused per-cloud reduction
  const int cloud = pt0 >> 10;  // 1024 points per cloud, 64 | 1024
  for (int nt = 0; nt < 24; ++nt) {
    const int n0 = nt * 16;
    v8f c = bias_init(globB, n0);
#pragma unroll
    for (int k0 = 0; k0 < 256; k0 += 32)
      c = wmma_f16(load_a(myAgg, 256, k0), load_b(wGlob, 256, n0, k0), c);
    float s = (c[0] + c[1]) + (c[2] + c[3]) + ((c[4] + c[5]) + (c[6] + c[7]));
    atomicAdd(&pooled[cloud * 384 + n0 + (lane & 15)], s);
  }
}

// ---------------------------------------------------------------------------
// Kernel 6: decoder MLP on 32 pooled rows (f32, single workgroup)
// ---------------------------------------------------------------------------
__global__ void decoder_kernel(const float* pooled,
                               const float* d1w, const float* d1b,
                               const float* d2w, const float* d2b,
                               const float* d3w, const float* d3b, float* out) {
  __shared__ float h1[32 * 256];
  __shared__ float h2[32 * 128];
  const int t = threadIdx.x;
  for (int idx = t; idx < 32 * 256; idx += 256) {
    const int cl = idx >> 8, j = idx & 255;
    float acc = 0.0f;
    for (int k = 0; k < 384; k += 4) {
      float4 p = *(const float4*)(pooled + cl * 384 + k);
      float4 w = *(const float4*)(d1w + j * 384 + k);
      acc += p.x * w.x + p.y * w.y + p.z * w.z + p.w * w.w;
    }
    h1[idx] = fast_tanh(acc * (1.0f / (float)PTS) + d1b[j]);  // mean folded in
  }
  __syncthreads();
  for (int idx = t; idx < 32 * 128; idx += 256) {
    const int cl = idx >> 7, j = idx & 127;
    float acc = 0.0f;
    for (int k = 0; k < 256; k += 4) {
      float4 p = *(const float4*)(h1 + cl * 256 + k);
      float4 w = *(const float4*)(d2w + j * 256 + k);
      acc += p.x * w.x + p.y * w.y + p.z * w.z + p.w * w.w;
    }
    h2[idx] = fast_tanh(acc + d2b[j]);
  }
  __syncthreads();
  for (int idx = t; idx < 32 * 16; idx += 256) {
    const int cl = idx >> 4, j = idx & 15;
    float acc = 0.0f;
    for (int k = 0; k < 128; k += 4) {
      float4 p = *(const float4*)(h2 + cl * 128 + k);
      float4 w = *(const float4*)(d3w + j * 128 + k);
      acc += p.x * w.x + p.y * w.y + p.z * w.z + p.w * w.w;
    }
    out[idx] = acc + d3b[j];
  }
}

// ---------------------------------------------------------------------------
extern "C" void kernel_launch(void* const* d_in, const int* in_sizes, int n_in,
                              void* d_out, int out_size, void* d_ws, size_t ws_size,
                              hipStream_t stream) {
  const float*     pos    = (const float*)d_in[0];
  const long long* eidx   = (const long long*)d_in[1];
  const float*     enc1_w = (const float*)d_in[2];
  const float*     enc1_b = (const float*)d_in[3];
  const float*     enc2_w = (const float*)d_in[4];
  const float*     enc2_b = (const float*)d_in[5];
  const float*     enc3_w = (const float*)d_in[6];
  const float*     enc3_b = (const float*)d_in[7];
  const float*     loc_w  = (const float*)d_in[8];
  const float*     loc_b  = (const float*)d_in[9];
  const float*     glob_w = (const float*)d_in[10];
  const float*     glob_b = (const float*)d_in[11];
  const float*     dec1_w = (const float*)d_in[12];
  const float*     dec1_b = (const float*)d_in[13];
  const float*     dec2_w = (const float*)d_in[14];
  const float*     dec2_b = (const float*)d_in[15];
  const float*     dec3_w = (const float*)d_in[16];
  const float*     dec3_b = (const float*)d_in[17];
  float* out = (float*)d_out;

  // Workspace carve-up (all segments 256B-aligned by construction)
  char* w = (char*)d_ws;
  _Float16* wE1   = (_Float16*)w; w += 64  * 32  * sizeof(_Float16);
  _Float16* wE2   = (_Float16*)w; w += 64  * 64  * sizeof(_Float16);
  _Float16* wE3   = (_Float16*)w; w += 128 * 64  * sizeof(_Float16);
  _Float16* wLoc  = (_Float16*)w; w += 256 * 160 * sizeof(_Float16);
  _Float16* wGlob = (_Float16*)w; w += 384 * 256 * sizeof(_Float16);
  _Float16* xEnc  = (_Float16*)w; w += (size_t)NPTS * 128 * sizeof(_Float16);
  float*    S     = (float*)w;    w += (size_t)NPTS * 132 * sizeof(float);
  _Float16* Sh    = (_Float16*)w; w += (size_t)NPTS * 160 * sizeof(_Float16);
  float*    pooled= (float*)w;    w += 32 * 384 * sizeof(float);

  // 0) weight conversion + pooled zeroing
  prep_kernel<<<dim3(384, 6), 256, 0, stream>>>(enc1_w, enc2_w, enc3_w, loc_w,
                                                glob_w, wE1, wE2, wE3, wLoc,
                                                wGlob, pooled);
  // 1) encoder (WMMA)
  encoder_kernel<<<NPTS / 64, 128, 0, stream>>>(pos, enc1_b, enc2_b, enc3_b,
                                                wE1, wE2, wE3, xEnc);
  // 2) self-loop init of the segment accumulator
  initS_kernel<<<(NPTS * 33 + 255) / 256, 256, 0, stream>>>(xEnc, S);
  // 3) edge scatter-add
  scatter_kernel<<<EEDGES / 8, 256, 0, stream>>>(eidx, pos, xEnc, S);
  // 4) scatter-mean normalize -> f16 padded
  norm_kernel<<<(NPTS * 40 + 255) / 256, 256, 0, stream>>>(S, Sh);
  // 5) local_nn-on-sums + global_nn + per-cloud pooling (WMMA)
  aggpool_kernel<<<NPTS / 64, 128, 0, stream>>>(Sh, wLoc, loc_b, wGlob, glob_b,
                                                pooled);
  // 6) decoder
  decoder_kernel<<<1, 256, 0, stream>>>(pooled, dec1_w, dec1_b, dec2_w, dec2_b,
                                        dec3_w, dec3_b, out);
}
